// KMeans_30459908063313
// MI455X (gfx1250) — matrix operand order repaired and checked by reading, not compiled, not run on hardware
//
#include <hip/hip_runtime.h>

typedef __attribute__((ext_vector_type(16))) __bf16 v16bf;
typedef __attribute__((ext_vector_type(8)))  float  v8f;

#define N_ROWS 65536
#define DIM    512
#define K_CENT 4096
#define KCHUNKS (K_CENT / 16)

// ---------------------------------------------------------------------------
// Prep: center[K,D] f32 -> bf16 (row-major, 4MB) + c_sq[K] f32.
// One wave (32 lanes) per center row; 16 elements per lane.
// ---------------------------------------------------------------------------
__global__ __launch_bounds__(32)
void kmeans_prep(const float* __restrict__ center,
                 __bf16* __restrict__ cb,
                 float* __restrict__ csq) {
    const int k    = blockIdx.x;
    const int lane = threadIdx.x;
    const float*  row  = center + (size_t)k * DIM;
    __bf16*       orow = cb + (size_t)k * DIM;
    float acc = 0.0f;
#pragma unroll
    for (int t = 0; t < 4; ++t) {
        const int base = t * 128 + lane * 4;
        float4 v = *(const float4*)(row + base);
        acc = fmaf(v.x, v.x, acc);
        acc = fmaf(v.y, v.y, acc);
        acc = fmaf(v.z, v.z, acc);
        acc = fmaf(v.w, v.w, acc);
        orow[base + 0] = (__bf16)v.x;
        orow[base + 1] = (__bf16)v.y;
        orow[base + 2] = (__bf16)v.z;
        orow[base + 3] = (__bf16)v.w;
    }
#pragma unroll
    for (int off = 16; off; off >>= 1) acc += __shfl_xor(acc, off, 32);
    if (lane == 0) csq[k] = acc;
}

// ---------------------------------------------------------------------------
// Main: each wave handles 16 rows of x vs all 4096 centers.
// score[n,k] = c_sq[k] - 2 * dot(x_n, c_k); x_sq is argmin-invariant.
// ---------------------------------------------------------------------------
__global__ __launch_bounds__(256)
void kmeans_argmin(const float* __restrict__ x,
                   const __bf16* __restrict__ cb,
                   const float* __restrict__ csq,
                   int* __restrict__ out) {
    const int lane    = threadIdx.x & 31;
    const int wave    = threadIdx.x >> 5;
    const int rowBase = blockIdx.x * 128 + wave * 16;
    const int m       = lane & 15;   // A: row within tile; B: column (center)
    const int half    = lane >> 4;   // lane half selects K sub-range

    // --- Preload A fragments: 16 rows x 512 d, bf16, contiguous K layout ---
    // lane<16 : elems 0..15 -> K{0..15}
    // lane>=16: elems 0..15 -> K{16..31}
    const float* xrow = x + (size_t)(rowBase + m) * DIM;
    v16bf A[16];
#pragma unroll
    for (int dc = 0; dc < 16; ++dc) {
        const int o0 = dc * 32 + half * 16;
        const int o1 = o0 + 8;
        float4 f0 = *(const float4*)(xrow + o0);
        float4 f1 = *(const float4*)(xrow + o0 + 4);
        float4 f2 = *(const float4*)(xrow + o1);
        float4 f3 = *(const float4*)(xrow + o1 + 4);
        v16bf a;
        a[0]  = (__bf16)f0.x; a[1]  = (__bf16)f0.y;
        a[2]  = (__bf16)f0.z; a[3]  = (__bf16)f0.w;
        a[4]  = (__bf16)f1.x; a[5]  = (__bf16)f1.y;
        a[6]  = (__bf16)f1.z; a[7]  = (__bf16)f1.w;
        a[8]  = (__bf16)f2.x; a[9]  = (__bf16)f2.y;
        a[10] = (__bf16)f2.z; a[11] = (__bf16)f2.w;
        a[12] = (__bf16)f3.x; a[13] = (__bf16)f3.y;
        a[14] = (__bf16)f3.z; a[15] = (__bf16)f3.w;
        A[dc] = a;
    }

    float best[8];
    int   bidx[8];
#pragma unroll
    for (int r = 0; r < 8; ++r) { best[r] = 3.4e38f; bidx[r] = 0; }

    for (int ct = 0; ct < KCHUNKS; ++ct) {
        const int kcol = ct * 16 + m;              // this lane's center column
        // B layout: lane = column, elems 0..15 -> K = half*16 + 0..15
        const __bf16* crow = cb + (size_t)kcol * DIM + half * 16;
        if (ct + 1 < KCHUNKS)  // WGP-scope prefetch of next chunk's B tile
            __builtin_prefetch(crow + (size_t)16 * DIM, 0, 3);

        // Two independent accumulator chains (even/odd D-chunks) for ILP.
        v8f acc0 = {};
        v8f acc1 = {};
#pragma unroll
        for (int dc = 0; dc < 16; dc += 2) {
            v16bf b0 = *(const v16bf*)(crow + dc * 32);        // 32B: 2x b128
            v16bf b1 = *(const v16bf*)(crow + dc * 32 + 32);
            acc0 = __builtin_amdgcn_wmma_f32_16x16x32_bf16(
                false, A[dc],     false, b0, (short)0, acc0, false, false);
            acc1 = __builtin_amdgcn_wmma_f32_16x16x32_bf16(
                false, A[dc + 1], false, b1, (short)0, acc1, false, false);
        }

        const float cs = csq[kcol];
#pragma unroll
        for (int r = 0; r < 8; ++r) {
            const float dot = acc0[r] + acc1[r];
            const float s   = fmaf(dot, -2.0f, cs);
            if (s < best[r]) { best[r] = s; bidx[r] = kcol; }  // first-k wins ties
        }
    }

    // Cross-lane argmin within each 16-lane half (xor masks stay in-half).
    // C layout: lanes 0-15 slot r -> row r; lanes 16-31 slot r -> row r+8.
#pragma unroll
    for (int off = 1; off < 16; off <<= 1) {
#pragma unroll
        for (int r = 0; r < 8; ++r) {
            const float ov = __shfl_xor(best[r], off, 32);
            const int   oi = __shfl_xor(bidx[r], off, 32);
            if (ov < best[r] || (ov == best[r] && oi < bidx[r])) {
                best[r] = ov; bidx[r] = oi;
            }
        }
    }
    if (lane == 0) {
#pragma unroll
        for (int r = 0; r < 8; ++r) out[rowBase + r] = bidx[r];
    } else if (lane == 16) {
#pragma unroll
        for (int r = 0; r < 8; ++r) out[rowBase + 8 + r] = bidx[r];
    }
}

// ---------------------------------------------------------------------------
extern "C" void kernel_launch(void* const* d_in, const int* in_sizes, int n_in,
                              void* d_out, int out_size, void* d_ws, size_t ws_size,
                              hipStream_t stream) {
    const float* x      = (const float*)d_in[0];   // [65536, 512] f32
    const float* center = (const float*)d_in[1];   // [4096, 512] f32

    __bf16* cb  = (__bf16*)d_ws;                                   // 4 MB
    float*  csq = (float*)((char*)d_ws +
                           (size_t)K_CENT * DIM * sizeof(__bf16)); // 16 KB
    int* out = (int*)d_out;                                        // [65536] i32

    kmeans_prep<<<K_CENT, 32, 0, stream>>>(center, cb, csq);
    kmeans_argmin<<<N_ROWS / 128, 256, 0, stream>>>(x, cb, csq, out);
}
